// MHA_block_36979668418627
// MI455X (gfx1250) — matrix operand order repaired and checked by reading, not compiled, not run on hardware
//
#include <hip/hip_runtime.h>
#include <hip/hip_bf16.h>
#include <math.h>

// Problem constants (from reference)
#define BB   2
#define SS   2048
#define DDIM 1024
#define HH   16
#define DKV  64
#define MROWS (BB*SS)        // 4096

typedef __attribute__((ext_vector_type(16))) _Float16 v16h;
typedef __attribute__((ext_vector_type(8)))  _Float16 h8;
typedef __attribute__((ext_vector_type(8)))  float    v8f;

// ---------------------------------------------------------------------------
// Async global->LDS staging (CDNA5 GLOBAL_LOAD_ASYNC_TO_LDS_B128, ASYNCcnt).
// The builtin exists on this toolchain; its parameters are pointers to
// int __attribute__((vector_size(16))) in AS1 (global) / AS3 (LDS).
// Generic pointers to __shared__ carry the LDS byte offset in their low 32
// bits (LDS aperture: addr[31:0] == LDS offset), so integer casts are safe.
// ---------------------------------------------------------------------------
typedef int v4i_vs __attribute__((vector_size(16)));
typedef __attribute__((address_space(1))) v4i_vs as1_v4i;
typedef __attribute__((address_space(3))) v4i_vs as3_v4i;

#if __has_builtin(__builtin_amdgcn_global_load_async_to_lds_b128)
#define USE_ASYNC 1
#else
#define USE_ASYNC 0
#endif

static __device__ inline void async_ld_b128(const _Float16* g, _Float16* l) {
#if USE_ASYNC
  __builtin_amdgcn_global_load_async_to_lds_b128(
      (as1_v4i*)(unsigned long long)g,
      (as3_v4i*)(unsigned)(unsigned long long)l,
      0, 0);
#else
  *(h8*)l = *(const h8*)g;
#endif
}
static __device__ inline void async_wait_all() {
#if USE_ASYNC
#if __has_builtin(__builtin_amdgcn_s_wait_asynccnt)
  __builtin_amdgcn_s_wait_asynccnt(0);
#else
  asm volatile("s_wait_asynccnt 0x0" ::: "memory");
#endif
#endif
}

// ---------------------------------------------------------------------------
// WMMA fragment helpers (CDNA5 16x16x32 f16, wave32).
// A-layout (16x32, MxK): lane l -> M=l&15; kb = (l<16)?0:8;
//   halves 0..7 = K=kb..kb+7, halves 8..15 = K=kb+16..kb+23.
// B-layout (32x16, KxN): lane l -> N=l&15; same K mapping as A (kb, kb+16).
// C/D (16x16 f32): reg r, lane l -> M = r + ((l<16)?0:8), N = l&15.
// ---------------------------------------------------------------------------
static __device__ inline v16h frag_a(const _Float16* rowbase, int kb) {
  h8 lo = *(const h8*)(rowbase + kb);
  h8 hi = *(const h8*)(rowbase + kb + 16);
  v16h r;
#pragma unroll
  for (int i = 0; i < 8; ++i) { r[i] = lo[i]; r[8 + i] = hi[i]; }
  return r;
}
static __device__ inline v16h frag_b(const _Float16* p) {
  h8 lo = *(const h8*)(p);
  h8 hi = *(const h8*)(p + 16);
  v16h r;
#pragma unroll
  for (int i = 0; i < 8; ++i) { r[i] = lo[i]; r[8 + i] = hi[i]; }
  return r;
}
static __device__ inline v8f wmma_f16(v16h a, v16h b, v8f c) {
  return __builtin_amdgcn_wmma_f32_16x16x32_f16(false, a, false, b,
                                                (short)0, c, false, false);
}
static __device__ inline v8f v8f_zero() {
  v8f z = {0.f, 0.f, 0.f, 0.f, 0.f, 0.f, 0.f, 0.f};
  return z;
}

// ---------------------------------------------------------------------------
// 1) LayerNorm (ddof=1, eps added to std) + cast to f16. One 256-thr WG / row.
// ---------------------------------------------------------------------------
__global__ __launch_bounds__(256) void ln_kernel(const float* __restrict__ x,
                                                 _Float16* __restrict__ xn) {
  __shared__ float r1[256];
  __shared__ float r2[256];
  const int row = blockIdx.x;
  const float* xr = x + (size_t)row * DDIM;
  float s = 0.f, s2 = 0.f;
  for (int c = threadIdx.x; c < DDIM; c += 256) {
    float v = xr[c];
    s += v; s2 += v * v;
  }
  r1[threadIdx.x] = s; r2[threadIdx.x] = s2;
  __syncthreads();
  for (int off = 128; off > 0; off >>= 1) {
    if ((int)threadIdx.x < off) {
      r1[threadIdx.x] += r1[threadIdx.x + off];
      r2[threadIdx.x] += r2[threadIdx.x + off];
    }
    __syncthreads();
  }
  float mean = r1[0] * (1.0f / DDIM);
  float var  = (r2[0] - (float)DDIM * mean * mean) * (1.0f / (DDIM - 1));
  float inv  = 1.0f / (sqrtf(var) + 1e-6f);
  _Float16* o = xn + (size_t)row * DDIM;
  for (int c = threadIdx.x; c < DDIM; c += 256)
    o[c] = (_Float16)((xr[c] - mean) * inv);
}

// ---------------------------------------------------------------------------
// 2) f32 -> f16 cast (weights).
// ---------------------------------------------------------------------------
__global__ void cast_f16(const float* __restrict__ in, _Float16* __restrict__ out,
                         int n) {
  int i = blockIdx.x * 256 + threadIdx.x;
  if (i < n) out[i] = (_Float16)in[i];
}

// ---------------------------------------------------------------------------
// 3) Tiled WMMA GEMM: C[MxN] = A[MxK](f16) * B[KxN](f16) (+ residual, f32 out)
//    256 thr = 8 waves; tile 128(M) x 128(N); wave tile 32x64; K-step 32.
//    Double-buffered LDS; A staged via async global->LDS, B staged through
//    registers (needs a transpose scatter) while WMMAs of the current slice
//    execute. LDS: A row-major [128][32], B transposed [128 n][32 k].
// ---------------------------------------------------------------------------
template <bool RES>
__global__ __launch_bounds__(256) void gemm_wmma(const _Float16* __restrict__ A,
                                                 const _Float16* __restrict__ Bm,
                                                 float* __restrict__ C,
                                                 const float* __restrict__ res,
                                                 int M, int N, int K) {
  __shared__ __align__(16) _Float16 lA[2][128 * 32];
  __shared__ __align__(16) _Float16 lBt[2][128 * 32];

  const int tid  = threadIdx.x;
  const int wave = tid >> 5;
  const int lane = tid & 31;
  const int half = lane >> 4;
  const int nidx = lane & 15;
  const int kbA  = half * 8;
  const int kbB  = half * 16;
  const int wr   = wave >> 1;        // 0..3 -> M stripe of 32
  const int wc   = wave & 1;         // 0..1 -> N stripe of 64
  const int m0   = blockIdx.y * 128;
  const int n0   = blockIdx.x * 128;
  // staging coords: A 128x32 (16 halves/thr), B 32x128 (16 halves/thr)
  const int ar = tid >> 1, ac = (tid & 1) * 16;
  const int br = tid >> 3, bc = (tid & 7) * 16;

  v8f acc[2][4];
#pragma unroll
  for (int i = 0; i < 2; ++i)
#pragma unroll
    for (int t = 0; t < 4; ++t) acc[i][t] = v8f_zero();

  { // prologue: stage K-slice 0 into buffer 0
    const _Float16* ga = A + (size_t)(m0 + ar) * K + ac;
    async_ld_b128(ga,     &lA[0][ar * 32 + ac]);
    async_ld_b128(ga + 8, &lA[0][ar * 32 + ac + 8]);
    const _Float16* gb = Bm + (size_t)br * N + n0 + bc;
    h8 p0 = *(const h8*)gb;
    h8 p1 = *(const h8*)(gb + 8);
#pragma unroll
    for (int j = 0; j < 8; ++j) {
      lBt[0][(bc + j) * 32 + br]     = p0[j];
      lBt[0][(bc + 8 + j) * 32 + br] = p1[j];
    }
  }
  async_wait_all();
  __syncthreads();

  for (int k0 = 0; k0 < K; k0 += 32) {
    const int cur = (k0 >> 5) & 1;
    const int nxt = cur ^ 1;
    const bool pre = (k0 + 32) < K;
    h8 pb0 = {0, 0, 0, 0, 0, 0, 0, 0}, pb1 = pb0;
    if (pre) { // issue next-slice loads before the math
      const _Float16* ga = A + (size_t)(m0 + ar) * K + (k0 + 32) + ac;
      async_ld_b128(ga,     &lA[nxt][ar * 32 + ac]);
      async_ld_b128(ga + 8, &lA[nxt][ar * 32 + ac + 8]);
      const _Float16* gb = Bm + (size_t)(k0 + 32 + br) * N + n0 + bc;
      pb0 = *(const h8*)gb;
      pb1 = *(const h8*)(gb + 8);
    }

#pragma unroll
    for (int i = 0; i < 2; ++i) {
      v16h af = frag_a(&lA[cur][(wr * 32 + i * 16 + nidx) * 32], kbA);
#pragma unroll
      for (int t = 0; t < 4; ++t) {
        v16h bf = frag_b(&lBt[cur][(wc * 64 + t * 16 + nidx) * 32 + kbA]);
        acc[i][t] = wmma_f16(af, bf, acc[i][t]);
      }
    }

    if (pre) { // transpose-scatter B into the next buffer
#pragma unroll
      for (int j = 0; j < 8; ++j) {
        lBt[nxt][(bc + j) * 32 + br]     = pb0[j];
        lBt[nxt][(bc + 8 + j) * 32 + br] = pb1[j];
      }
      async_wait_all();
    }
    __syncthreads();
  }

#pragma unroll
  for (int i = 0; i < 2; ++i)
#pragma unroll
    for (int t = 0; t < 4; ++t)
#pragma unroll
      for (int r = 0; r < 8; ++r) {
        int row = m0 + wr * 32 + i * 16 + r + half * 8;
        int col = n0 + wc * 64 + t * 16 + nidx;
        float v = acc[i][t][r];
        if (RES) v += res[(size_t)row * N + col];
        C[(size_t)row * N + col] = v;
      }
}

// ---------------------------------------------------------------------------
// 4) RoPE (interleaved rotate-half) + pack q,k,v into [B,H,S,64] f16.
// ---------------------------------------------------------------------------
__global__ void rope_pack(const float* __restrict__ q, const float* __restrict__ k,
                          const float* __restrict__ v, _Float16* __restrict__ qh,
                          _Float16* __restrict__ kh, _Float16* __restrict__ vh) {
  size_t idx = (size_t)blockIdx.x * 256 + threadIdx.x;
  if (idx >= (size_t)BB * SS * HH * 32) return;
  int i = (int)(idx & 31);
  int h = (int)((idx >> 5) & 15);
  int s = (int)((idx >> 9) & 2047);
  int b = (int)(idx >> 20);

  float inv = __expf((float)(2 * i) * (-9.210340371976184f / 64.0f));
  float ang = (float)s * inv;
  float cs = __cosf(ang), sn = __sinf(ang);

  size_t src = ((size_t)(b * SS + s)) * DDIM + h * DKV + 2 * i;
  size_t dst = ((size_t)((b * HH + h) * SS + s)) * DKV + 2 * i;

  float q0 = q[src], q1 = q[src + 1];
  float k0 = k[src], k1 = k[src + 1];
  qh[dst]     = (_Float16)(q0 * cs - q1 * sn);
  qh[dst + 1] = (_Float16)(q1 * cs + q0 * sn);
  kh[dst]     = (_Float16)(k0 * cs - k1 * sn);
  kh[dst + 1] = (_Float16)(k1 * cs + k0 * sn);
  vh[dst]     = (_Float16)v[src];
  vh[dst + 1] = (_Float16)v[src + 1];
}

// ---------------------------------------------------------------------------
// 5) Flash attention: grid (S/64, B*H), 128 thr = 4 waves; each wave owns a
//    16-row query stripe. Online softmax with 16-lane shuffle reductions.
//    P goes through wave-private LDS to convert C-layout -> A-layout.
//    Q/K staged via async global->LDS; V needs a transpose so stays manual.
// ---------------------------------------------------------------------------
__global__ __launch_bounds__(128) void flash_attn(const _Float16* __restrict__ qh,
                                                  const _Float16* __restrict__ kh,
                                                  const _Float16* __restrict__ vh,
                                                  const int* __restrict__ mask,
                                                  _Float16* __restrict__ atn) {
  __shared__ __align__(16) _Float16 lQ[64 * 64];
  __shared__ __align__(16) _Float16 lK[64 * 64];       // [key][d]
  __shared__ __align__(16) _Float16 lVt[64 * 64];      // [d][key]
  __shared__ __align__(16) _Float16 lP[4][16 * 64];    // per-wave P tile

  const int tid  = threadIdx.x;
  const int wave = tid >> 5;
  const int lane = tid & 31;
  const int half = lane >> 4;
  const int nidx = lane & 15;
  const int kbA  = half * 8;
  const int kbB  = half * 16;
  const int bh   = blockIdx.y;
  const int b    = bh >> 4;
  const int h    = bh & 15;
  const int q0   = blockIdx.x * 64;

  const _Float16* Qg = qh + ((size_t)bh * SS) * DKV;
  const _Float16* Kg = kh + ((size_t)bh * SS) * DKV;
  const _Float16* Vg = vh + ((size_t)bh * SS) * DKV;

  { // stage Q block once (async): thread row = tid/2, 32 halves at (tid&1)*32
    int r = tid >> 1, c = (tid & 1) * 32;
    const _Float16* g = Qg + (size_t)(q0 + r) * DKV + c;
#pragma unroll
    for (int j = 0; j < 4; ++j)
      async_ld_b128(g + 8 * j, &lQ[r * 64 + c + 8 * j]);
  }

  float mrow[8], lrow[8];
  v8f acc[4];
#pragma unroll
  for (int r = 0; r < 8; ++r) { mrow[r] = -1e30f; lrow[r] = 0.f; }
#pragma unroll
  for (int t = 0; t < 4; ++t) acc[t] = v8f_zero();

  for (int kb = 0; kb < SS; kb += 64) {
    { // stage K (async, row-major) and V (manual transpose)
      int r = tid >> 1, c = (tid & 1) * 32;
      const _Float16* gk = Kg + (size_t)(kb + r) * DKV + c;
#pragma unroll
      for (int j = 0; j < 4; ++j)
        async_ld_b128(gk + 8 * j, &lK[r * 64 + c + 8 * j]);
      const _Float16* gv = Vg + (size_t)(kb + r) * DKV + c;
#pragma unroll
      for (int j = 0; j < 4; ++j) {
        h8 p = *(const h8*)(gv + 8 * j);
#pragma unroll
        for (int e = 0; e < 8; ++e) lVt[(c + 8 * j + e) * 64 + r] = p[e];
      }
    }
    async_wait_all();
    __syncthreads();

    // S = Q * K^T  (per wave: 16 rows x 64 keys, d-dim 64 -> 2 k-steps)
    v8f sc[4];
#pragma unroll
    for (int t = 0; t < 4; ++t) sc[t] = v8f_zero();
    for (int d0 = 0; d0 < DKV; d0 += 32) {
      v16h aq = frag_a(&lQ[(wave * 16 + nidx) * 64 + d0], kbA);
#pragma unroll
      for (int t = 0; t < 4; ++t) {
        v16h bk = frag_b(&lK[(t * 16 + nidx) * 64 + d0 + kbA]);
        sc[t] = wmma_f16(aq, bk, sc[t]);
      }
    }

    // scale + mask
#pragma unroll
    for (int t = 0; t < 4; ++t)
#pragma unroll
      for (int r = 0; r < 8; ++r) {
        int qrow = q0 + wave * 16 + r + half * 8;
        int key  = kb + t * 16 + nidx;
        float sv = sc[t][r] * 0.125f;  // 1/sqrt(64)
        if (mask[((size_t)b * SS + qrow) * SS + key] == 0) sv = -1e9f;
        sc[t][r] = sv;
      }

    // online softmax per row (rows striped over 16-lane halves)
#pragma unroll
    for (int r = 0; r < 8; ++r) {
      float mx = sc[0][r];
#pragma unroll
      for (int t = 1; t < 4; ++t) mx = fmaxf(mx, sc[t][r]);
#pragma unroll
      for (int o = 1; o < 16; o <<= 1) mx = fmaxf(mx, __shfl_xor(mx, o, 32));
      float mnew  = fmaxf(mrow[r], mx);
      float alpha = __expf(mrow[r] - mnew);
      mrow[r] = mnew;
      float psum = 0.f;
#pragma unroll
      for (int t = 0; t < 4; ++t) {
        float p = __expf(sc[t][r] - mnew);
        sc[t][r] = p;
        psum += p;
      }
#pragma unroll
      for (int o = 1; o < 16; o <<= 1) psum += __shfl_xor(psum, o, 32);
      lrow[r] = lrow[r] * alpha + psum;
#pragma unroll
      for (int t = 0; t < 4; ++t) acc[t][r] *= alpha;
    }

    // P (C-layout) -> wave-private LDS -> A-layout fragments
#pragma unroll
    for (int t = 0; t < 4; ++t)
#pragma unroll
      for (int r = 0; r < 8; ++r)
        lP[wave][(r + half * 8) * 64 + t * 16 + nidx] = (_Float16)sc[t][r];

    // O += P * V  (k-dim = 64 keys -> 2 k-steps)
    for (int k0 = 0; k0 < 64; k0 += 32) {
      v16h ap = frag_a(&lP[wave][nidx * 64 + k0], kbA);
#pragma unroll
      for (int t = 0; t < 4; ++t) {
        v16h bv = frag_b(&lVt[(t * 16 + nidx) * 64 + k0 + kbA]);
        acc[t] = wmma_f16(ap, bv, acc[t]);
      }
    }
    __syncthreads();
  }

  // epilogue: O / l -> attn buffer [B,S,H*DV] f16
#pragma unroll
  for (int t = 0; t < 4; ++t)
#pragma unroll
    for (int r = 0; r < 8; ++r) {
      int qrow = q0 + wave * 16 + r + half * 8;
      int col  = h * DKV + t * 16 + nidx;
      atn[((size_t)(b * SS + qrow)) * DDIM + col] =
          (_Float16)(acc[t][r] / lrow[r]);
    }
}

// ---------------------------------------------------------------------------
// Launch
// ---------------------------------------------------------------------------
extern "C" void kernel_launch(void* const* d_in, const int* in_sizes, int n_in,
                              void* d_out, int out_size, void* d_ws,
                              size_t ws_size, hipStream_t stream) {
  const float* x    = (const float*)d_in[0];
  const int*   mask = (const int*)d_in[1];
  const float* Wq   = (const float*)d_in[2];
  const float* Wk   = (const float*)d_in[3];
  const float* Wv   = (const float*)d_in[4];
  const float* Wfc  = (const float*)d_in[5];

  char* ws = (char*)d_ws;
  const size_t MB = 1u << 20;
  _Float16* xn_h  = (_Float16*)(ws + 0 * MB);    // 8 MB  (4096x1024 f16)
  _Float16* wq_h  = (_Float16*)(ws + 8 * MB);    // 2 MB
  _Float16* wk_h  = (_Float16*)(ws + 10 * MB);   // 2 MB
  _Float16* wv_h  = (_Float16*)(ws + 12 * MB);   // 2 MB
  _Float16* wfc_h = (_Float16*)(ws + 14 * MB);   // 2 MB
  float*    qf    = (float*)(ws + 16 * MB);      // 16 MB
  float*    kf    = (float*)(ws + 32 * MB);      // 16 MB
  float*    vf    = (float*)(ws + 48 * MB);      // 16 MB
  _Float16* qh    = (_Float16*)(ws + 64 * MB);   // 8 MB
  _Float16* kh    = (_Float16*)(ws + 72 * MB);   // 8 MB
  _Float16* vh    = (_Float16*)(ws + 80 * MB);   // 8 MB
  _Float16* ath   = (_Float16*)(ws + 88 * MB);   // 8 MB  (total 96 MB)

  // 1) LayerNorm -> f16
  ln_kernel<<<MROWS, 256, 0, stream>>>(x, xn_h);

  // 2) weights -> f16
  const int wn = DDIM * DDIM;
  cast_f16<<<(wn + 255) / 256, 256, 0, stream>>>(Wq, wq_h, wn);
  cast_f16<<<(wn + 255) / 256, 256, 0, stream>>>(Wk, wk_h, wn);
  cast_f16<<<(wn + 255) / 256, 256, 0, stream>>>(Wv, wv_h, wn);
  cast_f16<<<(wn + 255) / 256, 256, 0, stream>>>(Wfc, wfc_h, wn);

  // 3) Q/K/V projections (WMMA, 128x128 tiles)
  dim3 gg(DDIM / 128, MROWS / 128), bb(256);
  gemm_wmma<false><<<gg, bb, 0, stream>>>(xn_h, wq_h, qf, nullptr, MROWS, DDIM, DDIM);
  gemm_wmma<false><<<gg, bb, 0, stream>>>(xn_h, wk_h, kf, nullptr, MROWS, DDIM, DDIM);
  gemm_wmma<false><<<gg, bb, 0, stream>>>(xn_h, wv_h, vf, nullptr, MROWS, DDIM, DDIM);

  // 4) RoPE + pack to [B,H,S,64] f16
  {
    size_t tot = (size_t)BB * SS * HH * 32;
    rope_pack<<<(unsigned)((tot + 255) / 256), 256, 0, stream>>>(qf, kf, vf, qh, kh, vh);
  }

  // 5) flash attention (WMMA)
  flash_attn<<<dim3(SS / 64, BB * HH), 128, 0, stream>>>(qh, kh, vh, mask, ath);

  // 6) output projection + residual (WMMA) -> f32 out
  gemm_wmma<true><<<gg, bb, 0, stream>>>(ath, wfc_h, (float*)d_out, x, MROWS, DDIM, DDIM);
}
